// TransformerLayer_62302795596152
// MI455X (gfx1250) — matrix operand order repaired and checked by reading, compiled, not run on hardware
//
#include <hip/hip_runtime.h>

// ============================================================================
// MI455X (gfx1250, wave32) implementation.
//
// Math note: reference applies Softmax over the head axis of size 1 -> attn
// is exactly 1.0, so q/k/Wq/Wk/x_g are dead and attention collapses to a
// token-sum of v, which folds into a reduction of Wv. Everything below is
// mathematically exact w.r.t. the reference (modulo f16 matrix-core rounding,
// which is free since the problem is HBM-bound: ~72MB traffic, ~17 GFLOP).
//
// Workspace usage: ~13.6 MB (see offsets in kernel_launch).
// ============================================================================

typedef __attribute__((ext_vector_type(16))) _Float16 v16h;
typedef __attribute__((ext_vector_type(8)))  float    v8f;

#define C2   2048
#define HW   32
#define CHW  65536      // 2048 * 32
#define KC   512        // K chunk staged in LDS per GEMM iteration
#define BSTR 528        // padded LDS stride in halves (16B-aligned, de-conflicted)

// ---------------------------------------------------------------------------
// Wvr[j, c] = sum_{l<32} Wv[(l*64+j)*2048 + c]   (j in [0,64))
// bvr[j]    = sum_{l<32} bv[l*64+j]
// ---------------------------------------------------------------------------
__global__ __launch_bounds__(256) void reduce_wv_kernel(
    const float* __restrict__ Wv, const float* __restrict__ bv,
    float* __restrict__ Wvr, float* __restrict__ bvr) {
  int gid = blockIdx.x * 256 + threadIdx.x;
  if (gid < 64 * C2) {
    int j = gid >> 11, c = gid & (C2 - 1);
    float s = 0.f;
#pragma unroll 4
    for (int l = 0; l < 32; ++l) s += Wv[(size_t)(l * 64 + j) * C2 + c];
    Wvr[gid] = s;                      // gid == j*2048 + c
  }
  if (gid < 64) {
    float s = 0.f;
    for (int l = 0; l < 32; ++l) s += bv[l * 64 + gid];
    bvr[gid] = s;
  }
}

// ---------------------------------------------------------------------------
// sv[n, j*32+s] = sum_c Wvr[j,c] * x[n,c,s] + bvr[j]
// One workgroup per query n; x[n] staged through LDS in 32KB chunks.
// ---------------------------------------------------------------------------
__global__ __launch_bounds__(256) void sv_kernel(
    const float* __restrict__ x, const float* __restrict__ Wvr,
    const float* __restrict__ bvr, float* __restrict__ sv) {
  __shared__ float xs[256 * 32];
  const int n = blockIdx.x, tid = threadIdx.x;
  const int s = tid & 31, jg = tid >> 5;          // jg in [0,8)
  float acc[8];
#pragma unroll
  for (int i = 0; i < 8; ++i) acc[i] = 0.f;
  for (int ck = 0; ck < 8; ++ck) {
    __syncthreads();
    for (int idx = tid; idx < 8192; idx += 256)
      xs[idx] = x[(size_t)n * CHW + ck * 8192 + idx];
    __syncthreads();
    for (int c = 0; c < 256; ++c) {
      float xv = xs[c * 32 + s];
      int cg = ck * 256 + c;
#pragma unroll
      for (int i = 0; i < 8; ++i)
        acc[i] += Wvr[(jg * 8 + i) * C2 + cg] * xv;
    }
  }
#pragma unroll
  for (int i = 0; i < 8; ++i) {
    int j = jg * 8 + i;
    sv[(size_t)n * C2 + j * 32 + s] = acc[i] + bvr[j];
  }
}

// ---------------------------------------------------------------------------
// Generic WMMA GEMM: D[m, col] = A[m,:] . B[:,col] over K=2048.
//  - 256 threads = 8 waves; each wave owns 16 rows (M) x 32 cols (two 16x16
//    accumulator tiles). Workgroup: 128 rows x 32 cols.
//  - B panel staged in LDS as f16 [col][k] with padded stride; A streamed
//    from HBM f32 and converted to f16 in registers (per-lane layout matches
//    the B fragment's contiguous k order, k = 16*(lane/16) + element).
//  mode 0 (AGG):  B = sv f32 [col*2048+k];       D: agg[n*2048+m] = 16*acc+16*bo[m]
//  mode 1 (MLP1): B = h1 f32 [cb*65536+k*32+col]; D: u[col*2048+m] = f16(relu(acc))
//  mode 2 (MLP2): B = u  f16 [col*2048+k];       D: h1[n*65536+m*32+hw] += acc
// ---------------------------------------------------------------------------
__global__ __launch_bounds__(256) void gemm_kernel(
    const float* __restrict__ A, const float* __restrict__ Bf32,
    const _Float16* __restrict__ Bf16, const float* __restrict__ bias,
    float* __restrict__ Dagg, _Float16* __restrict__ Du,
    float* __restrict__ Dh1, int mode) {
  __shared__ __attribute__((aligned(32))) _Float16 bsh[32 * BSTR];
  const int tid = threadIdx.x;
  const int wave = tid >> 5, lane = tid & 31;
  const int lane_m = lane & 15, hi = lane >> 4;
  const int rowbase = blockIdx.x * 128;
  const int cb = blockIdx.y;
  const int colbase = cb * 32;
  const int row = rowbase + wave * 16 + lane_m;
  const float* Arow = A + (size_t)row * C2;
  const int c0  = hi * 16;   // A-fragment K sub-offset (lanes 16-31 hold K+16)
  const int kb0 = hi * 16;   // B-fragment K sub-offset (lanes 16-31 hold K+16)

  v8f acc0 = {};
  v8f acc1 = {};

  for (int kc = 0; kc < C2; kc += KC) {
    __syncthreads();
    if (mode == 1) {                       // B[k,col] = h1[cb, k, col] (coalesced)
      for (int idx = tid; idx < 32 * KC; idx += 256) {
        int kk = idx >> 5, col = idx & 31;
        bsh[col * BSTR + kk] = (_Float16)Bf32[(size_t)cb * CHW + (size_t)kc * 32 + idx];
      }
    } else if (mode == 0) {                // B[k,col] = sv[col, k]
      for (int idx = tid; idx < 32 * KC; idx += 256) {
        int col = idx >> 9, kk = idx & (KC - 1);
        bsh[col * BSTR + kk] = (_Float16)Bf32[(size_t)col * C2 + kc + kk];
      }
    } else {                               // B[k,col] = u[colbase+col, k] (f16)
      for (int idx = tid; idx < 32 * KC; idx += 256) {
        int col = idx >> 9, kk = idx & (KC - 1);
        bsh[col * BSTR + kk] = Bf16[(size_t)(colbase + col) * C2 + kc + kk];
      }
    }
    __syncthreads();

    if (kc + KC < C2)                      // pull next A chunk toward the WGP
      __builtin_prefetch(Arow + kc + KC, 0, 1);

    for (int kk = 0; kk < KC; kk += 32) {
      const float* ap = Arow + kc + kk + c0;
      float4 a0 = *(const float4*)(ap);
      float4 a1 = *(const float4*)(ap + 4);
      float4 a2 = *(const float4*)(ap + 8);
      float4 a3 = *(const float4*)(ap + 12);
      v16h af;
      af[0]  = (_Float16)a0.x; af[1]  = (_Float16)a0.y;
      af[2]  = (_Float16)a0.z; af[3]  = (_Float16)a0.w;
      af[4]  = (_Float16)a1.x; af[5]  = (_Float16)a1.y;
      af[6]  = (_Float16)a1.z; af[7]  = (_Float16)a1.w;
      af[8]  = (_Float16)a2.x; af[9]  = (_Float16)a2.y;
      af[10] = (_Float16)a2.z; af[11] = (_Float16)a2.w;
      af[12] = (_Float16)a3.x; af[13] = (_Float16)a3.y;
      af[14] = (_Float16)a3.z; af[15] = (_Float16)a3.w;
      const v16h b0 = *(const v16h*)&bsh[lane_m * BSTR + kk + kb0];
      const v16h b1 = *(const v16h*)&bsh[(16 + lane_m) * BSTR + kk + kb0];
      acc0 = __builtin_amdgcn_wmma_f32_16x16x32_f16(false, af, false, b0,
                                                    (short)0, acc0, false, false);
      acc1 = __builtin_amdgcn_wmma_f32_16x16x32_f16(false, af, false, b1,
                                                    (short)0, acc1, false, false);
    }
  }

  // C/D layout: VGPR r -> lanes 0-15: (M=r, N=lane); lanes 16-31: (M=8+r, N=lane-16)
  const int m0 = rowbase + wave * 16 + hi * 8;
  if (mode == 0) {
#pragma unroll
    for (int r = 0; r < 8; ++r) {
      int m = m0 + r;
      float b16 = 16.f * bias[m];
      Dagg[(size_t)(colbase + lane_m) * C2 + m]      = 16.f * acc0[r] + b16;
      Dagg[(size_t)(colbase + 16 + lane_m) * C2 + m] = 16.f * acc1[r] + b16;
    }
  } else if (mode == 1) {
#pragma unroll
    for (int r = 0; r < 8; ++r) {
      int m = m0 + r;
      Du[(size_t)(colbase + lane_m) * C2 + m]      = (_Float16)fmaxf(acc0[r], 0.f);
      Du[(size_t)(colbase + 16 + lane_m) * C2 + m] = (_Float16)fmaxf(acc1[r], 0.f);
    }
  } else {
#pragma unroll
    for (int r = 0; r < 8; ++r) {
      int m = m0 + r;
      int col0 = colbase + lane_m, col1 = col0 + 16;
      size_t i0 = (size_t)(col0 >> 5) * CHW + (size_t)m * 32 + (col0 & 31);
      size_t i1 = (size_t)(col1 >> 5) * CHW + (size_t)m * 32 + (col1 & 31);
      Dh1[i0] += acc0[r];   // h1 <- h1 + h2 (each element touched exactly once)
      Dh1[i1] += acc1[r];
    }
  }
}

// ---------------------------------------------------------------------------
// LayerNorm over [C,H,W] (65536 elems) per query n, with per-element affine.
// mode 0: val = x[n,i] + agg[n, i>>5]  (agg broadcast over spatial) -> out=h1
// mode 1: val = src[n,i]                                            -> out
// ---------------------------------------------------------------------------
__global__ __launch_bounds__(256) void add_ln_kernel(
    const float* __restrict__ x, const float* __restrict__ agg,
    const float* __restrict__ src, const float* __restrict__ gamma,
    const float* __restrict__ beta, float* __restrict__ out, int mode) {
  __shared__ float s1[256], s2[256];
  const int n = blockIdx.x, tid = threadIdx.x;
  float sum = 0.f, sq = 0.f;
  for (int i = tid; i < CHW; i += 256) {
    float v = (mode == 0) ? (x[(size_t)n * CHW + i] + agg[n * C2 + (i >> 5)])
                          : src[(size_t)n * CHW + i];
    sum += v; sq += v * v;
  }
  s1[tid] = sum; s2[tid] = sq;
  __syncthreads();
  for (int off = 128; off > 0; off >>= 1) {
    if (tid < off) { s1[tid] += s1[tid + off]; s2[tid] += s2[tid + off]; }
    __syncthreads();
  }
  const float mu   = s1[0] * (1.f / CHW);
  const float var  = s2[0] * (1.f / CHW) - mu * mu;
  const float rstd = rsqrtf(var + 1e-5f);
  for (int i = tid; i < CHW; i += 256) {
    float v = (mode == 0) ? (x[(size_t)n * CHW + i] + agg[n * C2 + (i >> 5)])
                          : src[(size_t)n * CHW + i];
    out[(size_t)n * CHW + i] = (v - mu) * rstd * gamma[i] + beta[i];
  }
}

// ---------------------------------------------------------------------------
extern "C" void kernel_launch(void* const* d_in, const int* in_sizes, int n_in,
                              void* d_out, int out_size, void* d_ws, size_t ws_size,
                              hipStream_t stream) {
  (void)in_sizes; (void)n_in; (void)out_size; (void)ws_size;
  const float* x    = (const float*)d_in[0];
  // d_in[1..5] (x_g, Wq, bq, Wk, bk) are dead: softmax over size-1 head axis == 1
  const float* Wv   = (const float*)d_in[6];
  const float* bv   = (const float*)d_in[7];
  const float* Wo   = (const float*)d_in[8];
  const float* bo   = (const float*)d_in[9];
  const float* W1   = (const float*)d_in[10];
  const float* W2   = (const float*)d_in[11];
  const float* ln1w = (const float*)d_in[12];
  const float* ln1b = (const float*)d_in[13];
  const float* ln2w = (const float*)d_in[14];
  const float* ln2b = (const float*)d_in[15];

  float* ws  = (float*)d_ws;
  float* Wvr = ws;                 //  64*2048           = 131072 f32
  float* bvr = ws + 131072;        //  64 f32
  float* sv  = ws + 131136;        //  32*2048           = 65536 f32
  float* agg = ws + 196672;        //  32*2048           = 65536 f32
  float* h1  = ws + 262208;        //  32*65536          = 2097152 f32
  _Float16* u = (_Float16*)(ws + 2359360);  // 1024*2048 = 2097152 f16  (~13.6MB total)

  reduce_wv_kernel<<<512, 256, 0, stream>>>(Wv, bv, Wvr, bvr);
  sv_kernel<<<32, 256, 0, stream>>>(x, Wvr, bvr, sv);
  // agg[n,o] = 16*(Wo . sv[n]) + 16*bo[o]
  gemm_kernel<<<dim3(16, 1), 256, 0, stream>>>(Wo, sv, nullptr, bo,
                                               agg, nullptr, nullptr, 0);
  // h1 = LayerNorm(x + agg)
  add_ln_kernel<<<32, 256, 0, stream>>>(x, agg, nullptr, ln1w, ln1b, h1, 0);
  // u = relu(W1 . h1)   (f16 intermediate)
  gemm_kernel<<<dim3(16, 32), 256, 0, stream>>>(W1, h1, nullptr, nullptr,
                                                nullptr, u, nullptr, 1);
  // h1 += W2 . u        (in-place residual sum)
  gemm_kernel<<<dim3(16, 32), 256, 0, stream>>>(W2, nullptr, u, nullptr,
                                                nullptr, nullptr, h1, 2);
  // out = LayerNorm(h1)
  add_ln_kernel<<<32, 256, 0, stream>>>(nullptr, nullptr, h1, ln2w, ln2b,
                                        (float*)d_out, 1);
}
